// Heat1D_31842887532660
// MI455X (gfx1250) — matrix operand order repaired and checked by reading, not compiled, not run on hardware
//
#include <hip/hip_runtime.h>
#include <math.h>

typedef __attribute__((ext_vector_type(16))) _Float16 v16h;
typedef __attribute__((ext_vector_type(8)))  _Float16 v8h;
typedef __attribute__((ext_vector_type(8)))  float    v8f;
typedef __attribute__((ext_vector_type(4)))  int      v4i;

#define GLOBAL_AS __attribute__((address_space(1)))
#define LDS_AS    __attribute__((address_space(3)))

#define BDIM 8
#define TDIM 2048
#define CDIM 192
#define KT32 (TDIM / 32)      // 64 k-slabs of width 32
#define NCB  (CDIM / 64)      // 3 column blocks
#define PI_F 3.14159265358979323846f

// gfx1250 async global->LDS copy (ASYNCcnt-tracked). Confirmed present.
#if defined(__HIP_DEVICE_COMPILE__) && __has_builtin(__builtin_amdgcn_global_load_async_to_lds_b128)
#define USE_ASYNC_LDS 1
#else
#define USE_ASYNC_LDS 0
#endif

#if defined(__HIP_DEVICE_COMPILE__) && __has_builtin(__builtin_amdgcn_s_wait_asynccnt)
#define ASYNC_WAIT0() __builtin_amdgcn_s_wait_asynccnt(0)
#elif USE_ASYNC_LDS
#define ASYNC_WAIT0() asm volatile("s_wait_asynccnt 0x0" ::: "memory")
#else
#define ASYNC_WAIT0()
#endif

// B operand lives in global memory pre-swizzled into WMMA fragment images:
// per (b, k-slab, cblock) a 2048-f16 image; element (k in 0..31, n in 0..63)
// at ((n>>4)<<9) + (((n&15) + ((k>>4)<<4))<<4) + (k&15). Staging a tile is
// then a pure contiguous async copy.
__device__ __forceinline__ size_t bswz_index(int b, int t, int c) {
    int kb = t >> 5, k = t & 31;
    int cb = c >> 6, n = c & 63;
    int img = ((n >> 4) << 9) + (((n & 15) + ((k >> 4) << 4)) << 4) + (k & 15);
    return ((((size_t)b * KT32 + kb) * NCB + cb) << 11) + img;
}

// ---------------------------------------------------------------------------
// Kernel 1: fused depthwise conv -> grouped conv; u -> swizzled f16, z -> f32.
// ---------------------------------------------------------------------------
__global__ void fused_dwconv(const float* __restrict__ x,
                             const float* __restrict__ w1, const float* __restrict__ b1,
                             const float* __restrict__ w2, const float* __restrict__ b2,
                             _Float16* __restrict__ u16s, float* __restrict__ z32) {
    int idx = blockIdx.x * blockDim.x + threadIdx.x;
    if (idx >= BDIM * TDIM * CDIM) return;
    int c = idx % CDIM;
    int t = (idx / CDIM) % TDIM;
    int b = idx / (CDIM * TDIM);

    auto X = [&](int tt, int ch) -> float {
        return (tt < 0 || tt >= TDIM) ? 0.f : x[((size_t)b * TDIM + tt) * CDIM + ch];
    };
    auto H1 = [&](int tt, int ch) -> float {
        if (tt < 0 || tt >= TDIM) return 0.f;
        return b1[ch] + w1[ch*3+0]*X(tt-1,ch) + w1[ch*3+1]*X(tt,ch) + w1[ch*3+2]*X(tt+1,ch);
    };

    int cu = c >> 1;
    int oz = CDIM + c;
    int cz = oz >> 1;
    float u = b2[c]  + w2[c*3+0]*H1(t-1,cu) + w2[c*3+1]*H1(t,cu) + w2[c*3+2]*H1(t+1,cu);
    float z = b2[oz] + w2[oz*3+0]*H1(t-1,cz) + w2[oz*3+1]*H1(t,cz) + w2[oz*3+2]*H1(t+1,cz);
    u16s[bswz_index(b, t, c)] = (_Float16)u;
    z32[idx] = z;
}

// ---------------------------------------------------------------------------
// Kernel 2: DCT-II basis matrix + transpose (both GEMM passes read row-major).
// ---------------------------------------------------------------------------
__global__ void build_cos(_Float16* __restrict__ cos16, _Float16* __restrict__ cosT16) {
    int idx = blockIdx.x * blockDim.x + threadIdx.x;
    if (idx >= TDIM * TDIM) return;
    int n = idx / TDIM, t = idx % TDIM;
    float w = __cosf((float)n * (((float)t + 0.5f) / (float)TDIM) * PI_F)
              * sqrtf(2.0f / (float)TDIM);
    if (n == 0) w *= 0.70710678118654752f;
    _Float16 h = (_Float16)w;
    cos16[(size_t)n * TDIM + t] = h;
    cosT16[(size_t)t * TDIM + n] = h;
}

// ---------------------------------------------------------------------------
// Kernel 3/4: WMMA GEMM  out[2048 x 192] = A(2048x2048) x B(2048x192) / batch.
// 256 thr = 8 waves (4x2); block tile 256(M) x 64(N); K-chunk 64 (two 32-wide
// slab images) -> 16 v_wmma per wave per barrier. All staging is contiguous
// 16B async global->LDS; LDS double-buffered (80 KB total).
// ---------------------------------------------------------------------------
__global__ __launch_bounds__(256) void dct_gemm(
    const _Float16* __restrict__ Amat,   // [2048][2048] f16 row-major (m,k)
    const _Float16* __restrict__ Bsw,    // swizzled fragment images (see bswz_index)
    float* __restrict__ outF32,          // pass 2: plain [b][m][c] f32
    _Float16* __restrict__ outF16,       // pass 1: swizzled f16 (+ decay^k)
    const float* __restrict__ kvec)      // [C]
{
    __shared__ __align__(32) _Float16 sA[2][2 * 16 * 512]; // 2 buf x 2 slabs x 16KB
    __shared__ __align__(32) _Float16 sB[2][2 * 2048];     // 2 buf x 2 slabs x 4KB

    const int tid  = threadIdx.x;
    const int lane = tid & 31;
    const int wave = tid >> 5;
    const int wm   = wave >> 1;                 // 0..3
    const int wn   = wave & 1;                  // 0..1
    const int mBase = blockIdx.x * 256;
    const int cb    = blockIdx.y;               // column block (64 ch)
    const int bz    = blockIdx.z;

    v8f acc[4][2] = {};

    // stage one 32-wide k-slab pair into buffer p
    auto stage = [&](int kb2, int p) {
#pragma unroll
        for (int h = 0; h < 2; ++h) {
            const int ks = kb2 * 2 + h;
            // ---- A slab: 256x32 = 1024 x 16B chunks, 4 per thread ----------
#if USE_ASYNC_LDS
#pragma unroll
            for (int r = 0; r < 4; ++r) {
                int q = tid + 256 * r;
                int m = q >> 2, j = q & 3;
                const _Float16* src = Amat + (size_t)(mBase + m) * TDIM + ks * 32 + j * 8;
                int off = h * 8192 +
                          ((m >> 4) << 9) + (((m & 15) + ((j >> 1) << 4)) << 4) + ((j & 1) << 3);
                __builtin_amdgcn_global_load_async_to_lds_b128(
                    (GLOBAL_AS v4i*)(void*)const_cast<_Float16*>(src),
                    (LDS_AS v4i*)(void*)(&sA[p][off]), 0, 0);
            }
            // ---- B slab: pre-swizzled image, 256 x 16B chunks, 1/thread ----
            const _Float16* srcB =
                Bsw + ((((size_t)bz * KT32 + ks) * NCB + cb) << 11) + tid * 8;
            __builtin_amdgcn_global_load_async_to_lds_b128(
                (GLOBAL_AS v4i*)(void*)const_cast<_Float16*>(srcB),
                (LDS_AS v4i*)(void*)(&sB[p][h * 2048 + tid * 8]), 0, 0);
#else
            v8h tmpA[4];
            int offA[4];
#pragma unroll
            for (int r = 0; r < 4; ++r) {       // issue all loads first
                int q = tid + 256 * r;
                int m = q >> 2, j = q & 3;
                tmpA[r] = *(const v8h*)(Amat + (size_t)(mBase + m) * TDIM + ks * 32 + j * 8);
                offA[r] = h * 8192 +
                          ((m >> 4) << 9) + (((m & 15) + ((j >> 1) << 4)) << 4) + ((j & 1) << 3);
            }
            v8h tmpB = *(const v8h*)(Bsw + ((((size_t)bz * KT32 + ks) * NCB + cb) << 11) + tid * 8);
#pragma unroll
            for (int r = 0; r < 4; ++r)
                *(v8h*)(&sA[p][offA[r]]) = tmpA[r];
            *(v8h*)(&sB[p][h * 2048 + tid * 8]) = tmpB;
#endif
        }
    };

    stage(0, 0);
    int p = 0;
    const int NK2 = KT32 / 2;                   // 32 iterations of K-chunk 64
    for (int kb2 = 0; kb2 < NK2; ++kb2) {
        ASYNC_WAIT0();                          // my async copies for buf p done
        __syncthreads();                        // everyone's staging visible

        if (kb2 + 1 < NK2) stage(kb2 + 1, p ^ 1);   // overlaps WMMAs below

#pragma unroll
        for (int h = 0; h < 2; ++h) {
            v16h aF[4], bF[2];
#pragma unroll
            for (int i = 0; i < 4; ++i)
                aF[i] = *(const v16h*)(&sA[p][h * 8192 + ((wm * 4 + i) << 9) + (lane << 4)]);
#pragma unroll
            for (int j = 0; j < 2; ++j)
                bF[j] = *(const v16h*)(&sB[p][h * 2048 + ((wn * 2 + j) << 9) + (lane << 4)]);

#pragma unroll
            for (int i = 0; i < 4; ++i)
#pragma unroll
                for (int j = 0; j < 2; ++j)
                    acc[i][j] = __builtin_amdgcn_wmma_f32_16x16x32_f16(
                        false, aF[i], false, bF[j], (short)0, acc[i][j], false, false);
        }
        p ^= 1;
    }

    // ---- epilogue: C/D layout  VGPR r -> M = r + 8*(lane/16), N = lane%16 --
    const int half = lane >> 4;
    const int col  = lane & 15;
#pragma unroll
    for (int i = 0; i < 4; ++i) {
#pragma unroll
        for (int j = 0; j < 2; ++j) {
            int gn = cb * 64 + (wn * 2 + j) * 16 + col;
#pragma unroll
            for (int r = 0; r < 8; ++r) {
                int gm = mBase + (wm * 4 + i) * 16 + r + 8 * half;
                float vv = acc[i][j][r];
                if (outF16) {   // forward pass: decay[n]^k[c] = exp(-(pi n/T)^2 k[c]),
                                // stored pre-swizzled for GEMM2's B stager
                    float wf = PI_F * (float)gm / (float)TDIM;
                    float s  = __expf(-wf * wf * kvec[gn]);
                    outF16[bswz_index(bz, gm, gn)] = (_Float16)(vv * s);
                } else {
                    outF32[((size_t)bz * TDIM + gm) * CDIM + gn] = vv;
                }
            }
        }
    }
}

// ---------------------------------------------------------------------------
// Kernel 5: LayerNorm over C + SiLU(z) gate. One block per (b,t).
// ---------------------------------------------------------------------------
__global__ void ln_gate(const float* __restrict__ u2, const float* __restrict__ z32,
                        const float* __restrict__ gamma, const float* __restrict__ beta,
                        float* __restrict__ g32) {
    int bt = blockIdx.x;
    int c  = threadIdx.x;
    __shared__ float s1[256], s2[256];
    size_t base = (size_t)bt * CDIM;
    float v = (c < CDIM) ? u2[base + c] : 0.f;
    s1[c] = v; s2[c] = v * v;
    __syncthreads();
    for (int off = 128; off > 0; off >>= 1) {
        if (c < off) { s1[c] += s1[c + off]; s2[c] += s2[c + off]; }
        __syncthreads();
    }
    float mu  = s1[0] / (float)CDIM;
    float var = s2[0] / (float)CDIM - mu * mu;
    if (c < CDIM) {
        float zn = z32[base + c];
        float g  = (v - mu) * rsqrtf(var + 1e-5f) * gamma[c] + beta[c];
        g32[base + c] = g * (zn / (1.f + __expf(-zn)));
    }
}

// ---------------------------------------------------------------------------
// Kernel 6: final depthwise conv, output [b][t][c] f32.
// ---------------------------------------------------------------------------
__global__ void out_conv(const float* __restrict__ g32, const float* __restrict__ ow,
                         const float* __restrict__ ob, float* __restrict__ out) {
    int idx = blockIdx.x * blockDim.x + threadIdx.x;
    if (idx >= BDIM * TDIM * CDIM) return;
    int c = idx % CDIM;
    int t = (idx / CDIM) % TDIM;
    int b = idx / (CDIM * TDIM);
    auto G = [&](int tt) -> float {
        return (tt < 0 || tt >= TDIM) ? 0.f : g32[((size_t)b * TDIM + tt) * CDIM + c];
    };
    out[idx] = ob[c] + ow[c*3+0]*G(t-1) + ow[c*3+1]*G(t) + ow[c*3+2]*G(t+1);
}

// ---------------------------------------------------------------------------
extern "C" void kernel_launch(void* const* d_in, const int* in_sizes, int n_in,
                              void* d_out, int out_size, void* d_ws, size_t ws_size,
                              hipStream_t stream) {
    (void)in_sizes; (void)n_in; (void)out_size; (void)ws_size;
    const float* x     = (const float*)d_in[0];
    const float* w1    = (const float*)d_in[1];
    const float* b1    = (const float*)d_in[2];
    const float* w2    = (const float*)d_in[3];
    const float* b2    = (const float*)d_in[4];
    const float* gamma = (const float*)d_in[5];
    const float* beta  = (const float*)d_in[6];
    const float* ow    = (const float*)d_in[7];
    const float* ob    = (const float*)d_in[8];
    const float* kvec  = (const float*)d_in[9];
    float* out = (float*)d_out;

    // workspace layout (64 MiB total)
    char* ws = (char*)d_ws;
    const size_t MiB = 1024 * 1024;
    _Float16* cos16  = (_Float16*)(ws + 0  * MiB);  //  8 MiB  [n][t]
    _Float16* cosT16 = (_Float16*)(ws + 8  * MiB);  //  8 MiB  [t][n]
    _Float16* u16s   = (_Float16*)(ws + 16 * MiB);  //  6 MiB  swizzled
    float*    z32    = (float*)   (ws + 22 * MiB);  // 12 MiB
    _Float16* a16s   = (_Float16*)(ws + 34 * MiB);  //  6 MiB  swizzled
    float*    u2     = (float*)   (ws + 40 * MiB);  // 12 MiB
    float*    g32    = (float*)   (ws + 52 * MiB);  // 12 MiB

    const int total = BDIM * TDIM * CDIM;           // 3,145,728

    fused_dwconv<<<total / 256, 256, 0, stream>>>(x, w1, b1, w2, b2, u16s, z32);
    build_cos<<<(TDIM * TDIM) / 256, 256, 0, stream>>>(cos16, cosT16);

    dim3 gg(TDIM / 256, NCB, BDIM);                 // (8, 3, 8)
    // forward DCT + decay^k scaling (swizzled f16 out)
    dct_gemm<<<gg, 256, 0, stream>>>(cos16,  u16s, nullptr, a16s, kvec);
    // inverse DCT via pre-transposed basis (f32 out)
    dct_gemm<<<gg, 256, 0, stream>>>(cosT16, a16s, u2, nullptr, kvec);

    ln_gate<<<BDIM * TDIM, 256, 0, stream>>>(u2, z32, gamma, beta, g32);
    out_conv<<<total / 256, 256, 0, stream>>>(g32, ow, ob, out);
}
